// customGraphSAGE5_29154238005826
// MI455X (gfx1250) — matrix operand order repaired and checked by reading, not compiled, not run on hardware
//
#include <hip/hip_runtime.h>

typedef _Float16 f16;
typedef __attribute__((ext_vector_type(16))) _Float16 v16h;
typedef __attribute__((ext_vector_type(8)))  _Float16 v8h;
typedef __attribute__((ext_vector_type(8)))  float    v8f;

typedef __attribute__((ext_vector_type(4))) unsigned int tdm_u32x4;
typedef __attribute__((ext_vector_type(8))) int          tdm_i32x8;
typedef __attribute__((ext_vector_type(4))) int          tdm_i32x4;

#define G_GRAPHS 512
#define E_TOT    16384

// f16 weight buffer offsets in d_ws (elements)
#define OFF_M1W1 0        // 256x144
#define OFF_M1W2 36864    // 128x256
#define OFF_M1W3 69632    // 64x128
#define OFF_M2W1 77824    // 256x208
#define OFF_M2W2 131072   // 128x256
#define OFF_M2W3 163840   // 64x128

// ---- LDS layout (bytes) -------------------------------------------------
#define L_HL      0        // f16[32*64]
#define L_SRC     4096     // int[32]
#define L_DST     4224     // int[32]
#define L_DEG     4352     // f32[32]
#define L_EF      4480     // f16[32*16]
#define L_BL      5504     // f32[64]
#define L_B1F     5760     // f32[256]  m2 b1
#define L_B2F     6784     // f32[128]
#define L_B3F     7296     // f32[64]
#define L_W4F     7552     // f32[64]
#define L_B1M     7808     // f32[256]  m1 b1
#define L_B2M     8832     // f32[128]
#define L_B3M     9344     // f32[64]
#define L_W4M     9600     // f32[64]
#define L_LIN     9856     // f32[32]  lin_w[0..30], [31]=lin_b
#define L_PRED    9984     // f32[32]
#define L_NB      10112    // f32[32*30]
#define L_RED     13952    // f32[8*16*16]
#define L_W2L     22144    // f16[128*256]   (filled by TDM)
#define L_W3L     87680    // f16[64*128]    (filled by TDM)
#define L_PL      104064   // f16[32*256]
#define L_CL      120448   // f16[32*256]
#define L_UNI     136832
// phase0 union: Wlr f32[2*4096] @ +0 ; h32 f32[2048] @ +32768 ; hn f32[2048] @ +40960
// phase1 union: VUE f16[32*160] @ +0 ; h1m f16[32*256] @ +10240 ; h2m f16[32*128] @ +26624
// phase2 union: h2 tiles f16[8][16*128] @ +0
#define SMEM_TOTAL (136832 + 49152)

static __device__ __forceinline__ v16h mk16(v8h lo, v8h hi) {
  return __builtin_shufflevector(lo, hi, 0,1,2,3,4,5,6,7,8,9,10,11,12,13,14,15);
}
// A fragment: lane holds row m, chunks {k0+hh*8 .. +7} and {+16..+23}
static __device__ __forceinline__ v16h afrag(const f16* rowbase, int k0, int hh) {
  const f16* p = rowbase + k0 + hh*8;
  return mk16(*(const v8h*)p, *(const v8h*)(p + 16));
}
// B fragment: lane holds column n, chunks {k0+hh*8 .. +7} and {+16..+23}
static __device__ __forceinline__ v16h bfrag(const f16* rowbase, int k0, int hh) {
  const f16* p = rowbase + k0 + hh*8;
  return mk16(*(const v8h*)p, *(const v8h*)(p + 16));
}
static __device__ __forceinline__ v8f wmma16(v16h a, v16h b, v8f c) {
  return __builtin_amdgcn_wmma_f32_16x16x32_f16(false, a, false, b, (short)0, c, false, false);
}
// intra-wave LDS store->load handoff (per-wave, no workgroup barrier)
static __device__ __forceinline__ void wave_lds_sync() {
  __builtin_amdgcn_wave_barrier();
  asm volatile("s_wait_dscnt 0" ::: "memory");
  __builtin_amdgcn_wave_barrier();
}

// Tensor Data Mover: 1-D contiguous tile (n8 x 8-byte elements) global -> LDS.
// D# layout per CDNA5 ISA ch.8: group0 {count=1, lds_addr, global_addr, type=2},
// group1 {data_size=8B, tensor_dim0=tile_dim0=n8, dim1=tile_dim1=1, strides=n8}.
static __device__ __forceinline__ void tdm_load_lds_1d(unsigned lds_off, const void* gsrc, int n8) {
  unsigned long long ga = (unsigned long long)(uintptr_t)gsrc;
  tdm_u32x4 g0 = { 1u,                                   // count=1 (valid), user mode
                   lds_off,                              // lds_addr (bytes)
                   (unsigned)(ga & 0xFFFFFFFFull),       // global_addr[31:0]
                   (unsigned)((ga >> 32) & 0x01FFFFFFull) | (2u << 30) }; // addr[56:32] | type=2
  unsigned un8 = (unsigned)n8;
  tdm_i32x8 g1 = { (int)(3u << 16),                      // workgroup_mask=0, data_size=3 (8B)
                   (int)((un8 & 0xFFFFu) << 16),         // tensor_dim0[15:0]
                   (int)((un8 >> 16) | (1u << 16)),      // tensor_dim0[31:16] | tensor_dim1[15:0]=1
                   (int)((un8 & 0xFFFFu) << 16),         // tensor_dim1[31:16]=0 | tile_dim0
                   1,                                    // tile_dim1=1, tile_dim2=0
                   (int)un8,                             // tensor_dim0_stride[31:0]
                   (int)((un8 & 0xFFFFu) << 16),         // stride0[47:32]=0 | stride1[15:0]
                   0 };                                  // stride1[47:16]=0
  tdm_i32x4 z4 = { 0, 0, 0, 0 };
#if __has_include(<hip/amd_detail/amd_gfx1250_TDM.h>)
  tdm_i32x8 z8 = { 0, 0, 0, 0, 0, 0, 0, 0 };
  __builtin_amdgcn_tensor_load_to_lds(g0, g1, z4, z4, z8, 0);
#else
  __builtin_amdgcn_tensor_load_to_lds(g0, g1, z4, z4, 0);
#endif
}

__global__ void cvt_f32_to_f16(const float* __restrict__ s, f16* __restrict__ d, int n) {
  int i = blockIdx.x * blockDim.x + threadIdx.x;
  if (i < n) d[i] = (f16)s[i];
}

__global__ __launch_bounds__(256)
void graphsage_fused(const float* __restrict__ x, const int* __restrict__ eidx,
                     const float* __restrict__ efeat,
                     const float* __restrict__ Wl, const float* __restrict__ bl,
                     const float* __restrict__ Wr,
                     const float* __restrict__ lin_w, const float* __restrict__ lin_b,
                     const float* __restrict__ m1b1, const float* __restrict__ m1b2,
                     const float* __restrict__ m1b3, const float* __restrict__ m1w4,
                     const float* __restrict__ m1b4,
                     const float* __restrict__ m2b1, const float* __restrict__ m2b2,
                     const float* __restrict__ m2b3, const float* __restrict__ m2w4,
                     const float* __restrict__ m2b4,
                     const f16* __restrict__ wf16, float* __restrict__ out) {
  extern __shared__ __align__(16) char smem[];
  const int g   = blockIdx.x;
  const int tid = threadIdx.x;
  const int wid = tid >> 5;
  const int lane = tid & 31;
  const int ml = lane & 15;
  const int hh = lane >> 4;

  f16*   Hl    = (f16*)(smem + L_HL);
  int*   srcl  = (int*)(smem + L_SRC);
  int*   dstl  = (int*)(smem + L_DST);
  float* degS  = (float*)(smem + L_DEG);
  f16*   ef16  = (f16*)(smem + L_EF);
  float* blf   = (float*)(smem + L_BL);
  float* b1f   = (float*)(smem + L_B1F);
  float* b2f   = (float*)(smem + L_B2F);
  float* b3f   = (float*)(smem + L_B3F);
  float* w4f   = (float*)(smem + L_W4F);
  float* b1m   = (float*)(smem + L_B1M);
  float* b2m   = (float*)(smem + L_B2M);
  float* b3m   = (float*)(smem + L_B3M);
  float* w4m   = (float*)(smem + L_W4M);
  float* linS  = (float*)(smem + L_LIN);
  float* prednn= (float*)(smem + L_PRED);
  float* nbS   = (float*)(smem + L_NB);
  float* redS  = (float*)(smem + L_RED);
  f16*   W2l   = (f16*)(smem + L_W2L);
  f16*   W3l   = (f16*)(smem + L_W3L);
  f16*   Pl    = (f16*)(smem + L_PL);
  f16*   Cl    = (f16*)(smem + L_CL);
  char*  uni   = smem + L_UNI;

  // ---------------- Phase 0: stage data + GNN (f32 VALU) ----------------
  {
    float* WlrS = (float*)uni;
    float* h32  = (float*)(uni + 32768);
    float* hn   = (float*)(uni + 40960);

    // Async-DMA the MLP2 hot-loop weights into LDS; overlaps with the whole
    // GNN phase. Wave 0 owns TENSORcnt and drains it before the phase-2 barrier.
    if (wid == 0) {
      tdm_load_lds_1d(L_W2L, wf16 + OFF_M2W2, (128 * 256 * 2) / 8);
      tdm_load_lds_1d(L_W3L, wf16 + OFF_M2W3, (64 * 128 * 2) / 8);
    }
    if (tid < 32) {
      int e = g * 32 + tid;
      srcl[tid] = eidx[e] - g * 32;
      dstl[tid] = eidx[E_TOT + e] - g * 32;
    }
    for (int i = tid; i < 512; i += 256) ef16[i] = (f16)efeat[g * 512 + i];
    for (int i = tid; i < 4096; i += 256) { WlrS[i] = Wl[i]; WlrS[4096 + i] = Wr[i]; }
    if (tid < 64) {
      blf[tid] = bl[tid];
      b3f[tid] = m2b3[tid]; w4f[tid] = m2w4[tid];
      b3m[tid] = m1b3[tid]; w4m[tid] = m1w4[tid];
    }
    b1f[tid] = m2b1[tid];
    b1m[tid] = m1b1[tid];
    if (tid < 128) { b2f[tid] = m2b2[tid]; b2m[tid] = m1b2[tid]; }
    if (tid < 31) linS[tid] = lin_w[tid];
    if (tid == 31) linS[31] = lin_b[0];
    for (int i = tid; i < 2048; i += 256) h32[i] = x[g * 2048 + i];
    __syncthreads();

    if (tid < 32) {
      int c = 0;
      for (int e = 0; e < 32; ++e) c += (dstl[e] == tid) ? 1 : 0;
      degS[tid] = (float)(c ? c : 1);
    }
    __syncthreads();

    for (int rnd = 0; rnd < 3; ++rnd) {
      for (int i = tid; i < 2048; i += 256) {       // deterministic fixed-order aggregation
        int n = i >> 6, k = i & 63;
        float s = 0.f;
        for (int e = 0; e < 32; ++e) if (dstl[e] == n) s += h32[srcl[e] * 64 + k];
        hn[i] = s / degS[n];
      }
      __syncthreads();
      float vals[8];
#pragma unroll
      for (int ii = 0; ii < 8; ++ii) {
        int i = tid + ii * 256;
        int n = i >> 6, d = i & 63;
        float acc = blf[d];
        const float* wl = WlrS + d * 64;
        const float* wr = WlrS + 4096 + d * 64;
        const float* ha = hn + n * 64;
        const float* hb = h32 + n * 64;
#pragma unroll 8
        for (int k = 0; k < 64; ++k) acc += ha[k] * wl[k] + hb[k] * wr[k];
        vals[ii] = acc > 0.f ? acc : 0.f;
      }
      __syncthreads();
#pragma unroll
      for (int ii = 0; ii < 8; ++ii) h32[tid + ii * 256] = vals[ii];
      __syncthreads();
    }
    for (int i = tid; i < 2048; i += 256) Hl[i] = (f16)h32[i];
    __syncthreads();
  }

  // ---------------- Phase 1: P, C precompute + full MLP1 (WMMA) ----------------
  {
    f16* VUE = (f16*)uni;                 // 32 x 160 : [v | u | ef | 0-pad]
    f16* h1m = (f16*)(uni + 10240);       // 32 x 256
    f16* h2m = (f16*)(uni + 26624);       // 32 x 128

    for (int i = tid; i < 32 * 160; i += 256) {
      int e = i / 160, c = i - e * 160;
      f16 v;
      if (c < 64)       v = Hl[srcl[e] * 64 + c];
      else if (c < 128) v = Hl[dstl[e] * 64 + (c - 64)];
      else if (c < 144) v = ef16[e * 16 + (c - 128)];
      else              v = (f16)0.f;
      VUE[i] = v;
    }
    __syncthreads();

    // 96 wave-tiles: 32 for P = H @ W1z^T, 32 for C = VUE @ W1vue^T + b1, 32 for MLP1-L1
    for (int t = wid; t < 96; t += 8) {
      v8f acc = {};
      if (t < 32) {                                   // P: K=64 (z-columns 128..191 of m2_w1)
        int mt = t >> 4, nt = t & 15;
        const f16* arow = Hl + (mt * 16 + ml) * 64;
        const f16* brow = wf16 + OFF_M2W1 + (nt * 16 + ml) * 208 + 128;
#pragma unroll
        for (int kt = 0; kt < 2; ++kt)
          acc = wmma16(afrag(arow, kt * 32, hh), bfrag(brow, kt * 32, hh), acc);
#pragma unroll
        for (int r = 0; r < 8; ++r)
          Pl[(mt * 16 + hh * 8 + r) * 256 + nt * 16 + ml] = (f16)acc[r];
      } else if (t < 64) {                            // C: K=160 padded, cols {0..127, 192..207}
        int tt = t - 32, mt = tt >> 4, nt = tt & 15;
        const f16* arow = VUE + (mt * 16 + ml) * 160;
        const f16* brow = wf16 + OFF_M2W1 + (nt * 16 + ml) * 208;
#pragma unroll
        for (int kt = 0; kt < 5; ++kt) {
          v16h a = afrag(arow, kt * 32, hh);
          int kb0 = kt * 32 + hh * 8, kb1 = kb0 + 16;
          v8h blo = {}, bhi = {};
          if (kb0 < 144) blo = *(const v8h*)(brow + (kb0 < 128 ? kb0 : kb0 + 64));
          if (kb1 < 144) bhi = *(const v8h*)(brow + (kb1 < 128 ? kb1 : kb1 + 64));
          acc = wmma16(a, mk16(blo, bhi), acc);
        }
#pragma unroll
        for (int r = 0; r < 8; ++r) {
          int col = nt * 16 + ml;
          Cl[(mt * 16 + hh * 8 + r) * 256 + col] = (f16)(acc[r] + b1f[col]);
        }
      } else {                                        // MLP1 layer1: K=144 padded to 160
        int tt = t - 64, mt = tt >> 4, nt = tt & 15;
        const f16* arow = VUE + (mt * 16 + ml) * 160;
        const f16* brow = wf16 + OFF_M1W1 + (nt * 16 + ml) * 144;
#pragma unroll
        for (int kt = 0; kt < 5; ++kt) {
          v16h a = afrag(arow, kt * 32, hh);
          int kb0 = kt * 32 + hh * 8, kb1 = kb0 + 16;
          v8h blo = {}, bhi = {};
          if (kb0 < 144) blo = *(const v8h*)(brow + kb0);
          if (kb1 < 144) bhi = *(const v8h*)(brow + kb1);
          acc = wmma16(a, mk16(blo, bhi), acc);
        }
#pragma unroll
        for (int r = 0; r < 8; ++r) {
          int col = nt * 16 + ml;
          float v = acc[r] + b1m[col];
          h1m[(mt * 16 + hh * 8 + r) * 256 + col] = (f16)(v > 0.f ? v : 0.f);
        }
      }
    }
    __syncthreads();

    // MLP1 layer2: 32x128, K=256
    for (int t = wid; t < 16; t += 8) {
      int mt = t >> 3, nt = t & 7;
      const f16* arow = h1m + (mt * 16 + ml) * 256;
      const f16* brow = wf16 + OFF_M1W2 + (nt * 16 + ml) * 256;
      v8f acc = {};
#pragma unroll
      for (int kt = 0; kt < 8; ++kt)
        acc = wmma16(afrag(arow, kt * 32, hh), bfrag(brow, kt * 32, hh), acc);
#pragma unroll
      for (int r = 0; r < 8; ++r) {
        int col = nt * 16 + ml;
        float v = acc[r] + b2m[col];
        h2m[(mt * 16 + hh * 8 + r) * 128 + col] = (f16)(v > 0.f ? v : 0.f);
      }
    }
    __syncthreads();

    // MLP1 layer3 + layer4 dot: waves 0,1 handle 16 rows each
    if (wid < 2) {
      int mt = wid;
      v8f a3[4] = {};
      const f16* arow = h2m + (mt * 16 + ml) * 128;
#pragma unroll
      for (int kt = 0; kt < 4; ++kt) {
        v16h a = afrag(arow, kt * 32, hh);
#pragma unroll
        for (int nt = 0; nt < 4; ++nt)
          a3[nt] = wmma16(a, bfrag(wf16 + OFF_M1W3 + (nt * 16 + ml) * 128, kt * 32, hh), a3[nt]);
      }
      float p[8] = {0,0,0,0,0,0,0,0};
#pragma unroll
      for (int nt = 0; nt < 4; ++nt) {
        int col = nt * 16 + ml;
        float w = w4m[col], bb = b3m[col];
#pragma unroll
        for (int r = 0; r < 8; ++r) {
          float v = a3[nt][r] + bb; v = v > 0.f ? v : 0.f;
          p[r] += v * w;
        }
      }
      float* rw = redS + wid * 256;
#pragma unroll
      for (int r = 0; r < 8; ++r) rw[(hh * 8 + r) * 16 + ml] = p[r];
      wave_lds_sync();
      if (lane < 16) {
        float s = 0.f;
#pragma unroll
        for (int c = 0; c < 16; ++c) s += rw[lane * 16 + c];
        prednn[mt * 16 + lane] = s + m1b4[0];
      }
    }
    // Wave 0 drains its outstanding TDM transfers (W2l/W3l) before the barrier
    // publishes them to all waves for phase 2.
    if (wid == 0) __builtin_amdgcn_s_wait_tensorcnt(0);
    __syncthreads();
  }

  // ---------------- Phase 2: MLP2 over 960 rows (60 wave-tiles of 16) ----------------
  {
    f16* myh2 = (f16*)uni + wid * (16 * 128);
    float* rw = redS + wid * 256;

    for (int t = wid; t < 60; t += 8) {
      int row = t * 16 + ml;
      int e = row / 30, j = row - e * 30;
      int v = srcl[e], u = dstl[e];
      int a_ = v < u ? v : u;
      int b_ = v < u ? u : v;
      int zn = j; if (zn >= a_) zn++; if (zn >= b_) zn++;
      const f16* crow = Cl + e * 256;
      const f16* prow = Pl + zn * 256;

      // layer 2: K=256, h1 = relu(C[e]+P[zn]) built directly into A-fragments
      v8f acc[8] = {};
#pragma unroll
      for (int kt = 0; kt < 8; ++kt) {
        int kb = kt * 32 + hh * 8;
        v8h c0 = *(const v8h*)(crow + kb),      p0 = *(const v8h*)(prow + kb);
        v8h c1 = *(const v8h*)(crow + kb + 16), p1 = *(const v8h*)(prow + kb + 16);
        v8h s0 = c0 + p0, s1 = c1 + p1;
        v16h a;
#pragma unroll
        for (int i = 0; i < 8; ++i) {
          f16 x0 = s0[i]; a[i]     = x0 > (f16)0.f ? x0 : (f16)0.f;
          f16 x1 = s1[i]; a[i + 8] = x1 > (f16)0.f ? x1 : (f16)0.f;
        }
#pragma unroll
        for (int nt = 0; nt < 8; ++nt)
          acc[nt] = wmma16(a, bfrag(W2l + (nt * 16 + ml) * 256, kt * 32, hh), acc[nt]);
      }
#pragma unroll
      for (int nt = 0; nt < 8; ++nt) {
        int col = nt * 16 + ml; float bb = b2f[col];
#pragma unroll
        for (int r = 0; r < 8; ++r) {
          float vv = acc[nt][r] + bb;
          myh2[(hh * 8 + r) * 128 + col] = (f16)(vv > 0.f ? vv : 0.f);
        }
      }
      wave_lds_sync();

      // layer 3: K=128
      v8f a3[4] = {};
      const f16* arow = myh2 + ml * 128;
#pragma unroll
      for (int kt = 0; kt < 4; ++kt) {
        v16h a = afrag(arow, kt * 32, hh);
#pragma unroll
        for (int nt = 0; nt < 4; ++nt)
          a3[nt] = wmma16(a, bfrag(W3l + (nt * 16 + ml) * 128, kt * 32, hh), a3[nt]);
      }
      // layer 4 dot with w4 (deterministic LDS reduction)
      float p[8] = {0,0,0,0,0,0,0,0};
#pragma unroll
      for (int nt = 0; nt < 4; ++nt) {
        int col = nt * 16 + ml;
        float w = w4f[col], bb = b3f[col];
#pragma unroll
        for (int r = 0; r < 8; ++r) {
          float vv = a3[nt][r] + bb; vv = vv > 0.f ? vv : 0.f;
          p[r] += vv * w;
        }
      }
#pragma unroll
      for (int r = 0; r < 8; ++r) rw[(hh * 8 + r) * 16 + ml] = p[r];
      wave_lds_sync();
      if (lane < 16) {
        float s = 0.f;
#pragma unroll
        for (int c = 0; c < 16; ++c) s += rw[lane * 16 + c];
        nbS[e * 30 + j] = s + m2b4[0];    // lane<16 => ml==lane => this lane's e/j
      }
      wave_lds_sync();
    }
    __syncthreads();

    // final: out = lin_b + lin_w[0]*pred_nn + sum_j lin_w[1+j]*nb[j]
    if (tid < 32) {
      float s = linS[31] + linS[0] * prednn[tid];
#pragma unroll
      for (int j = 0; j < 30; ++j) s += linS[1 + j] * nbS[tid * 30 + j];
      out[g * 32 + tid] = s;
    }
  }
}

extern "C" void kernel_launch(void* const* d_in, const int* in_sizes, int n_in,
                              void* d_out, int out_size, void* d_ws, size_t ws_size,
                              hipStream_t stream) {
  const float* x     = (const float*)d_in[0];
  const int*   eidx  = (const int*)d_in[1];
  const float* efeat = (const float*)d_in[2];
  const float* Wl    = (const float*)d_in[5];
  const float* blv   = (const float*)d_in[6];
  const float* Wr    = (const float*)d_in[7];
  const float* lin_w = (const float*)d_in[8];
  const float* lin_b = (const float*)d_in[9];
  const float* m1w1 = (const float*)d_in[10]; const float* m1b1 = (const float*)d_in[11];
  const float* m1w2 = (const float*)d_in[12]; const float* m1b2 = (const float*)d_in[13];
  const float* m1w3 = (const float*)d_in[14]; const float* m1b3 = (const float*)d_in[15];
  const float* m1w4 = (const float*)d_in[16]; const float* m1b4 = (const float*)d_in[17];
  const float* m2w1 = (const float*)d_in[18]; const float* m2b1 = (const float*)d_in[19];
  const float* m2w2 = (const float*)d_in[20]; const float* m2b2 = (const float*)d_in[21];
  const float* m2w3 = (const float*)d_in[22]; const float* m2b3 = (const float*)d_in[23];
  const float* m2w4 = (const float*)d_in[24]; const float* m2b4 = (const float*)d_in[25];
  f16* wf16 = (f16*)d_ws;

  auto cvt = [&](const float* s, int off, int n) {
    cvt_f32_to_f16<<<(n + 255) / 256, 256, 0, stream>>>(s, wf16 + off, n);
  };
  cvt(m1w1, OFF_M1W1, 256 * 144);
  cvt(m1w2, OFF_M1W2, 128 * 256);
  cvt(m1w3, OFF_M1W3, 64 * 128);
  cvt(m2w1, OFF_M2W1, 256 * 208);
  cvt(m2w2, OFF_M2W2, 128 * 256);
  cvt(m2w3, OFF_M2W3, 64 * 128);

  graphsage_fused<<<G_GRAPHS, 256, SMEM_TOTAL, stream>>>(
      x, eidx, efeat, Wl, blv, Wr, lin_w, lin_b,
      m1b1, m1b2, m1b3, m1w4, m1b4,
      m2b1, m2b2, m2b3, m2w4, m2b4,
      wf16, (float*)d_out);
}
